// GPT_4879082848607
// MI455X (gfx1250) — matrix operand order repaired and checked by reading, not compiled, not run on hardware
//
#include <hip/hip_runtime.h>

#define Bn  8
#define Tn  1024
#define Cn  1024
#define Hn  16
#define HSn 64
#define NBn 63
#define BHn 128   // B*H

typedef __attribute__((ext_vector_type(16))) _Float16 v16h;
typedef __attribute__((ext_vector_type(8)))  float    v8f;

__device__ __forceinline__ v8f wmma16(v16h a, v16h b, v8f c){
  return __builtin_amdgcn_wmma_f32_16x16x32_f16(false, a, false, b, (short)0, c, false, false);
}
__device__ __forceinline__ v8f zero8(){ v8f z = {0.f,0.f,0.f,0.f,0.f,0.f,0.f,0.f}; return z; }

// Async copy of 32 contiguous bytes global -> LDS (2 x B128), tracked by ASYNCcnt.
// IOFFSET is added to BOTH the LDS and the global address (ISA 15.18.3 op 98).
__device__ __forceinline__ void async_copy32(unsigned lds_addr, const _Float16* g){
  unsigned long long ga = (unsigned long long)(size_t)g;
  asm volatile("global_load_async_to_lds_b128 %0, %1, off\n\t"
               "global_load_async_to_lds_b128 %0, %1, off offset:16"
               :: "v"(lds_addr), "v"(ga) : "memory");
}
__device__ __forceinline__ void async_wait0(){
  asm volatile("s_wait_asynccnt 0x0" ::: "memory");
}

// A fragment: 16(M) x 32(K) f16 from row-major [M x >=k0+32], leading dim ld.
// lanes 0-15: K = k0+0..15; lanes 16-31: K = k0+16..31
__device__ __forceinline__ v16h ldA16(const _Float16* base, int ld, int k0, int lane){
  int m = lane & 15; int kh = (lane >> 4) << 4;
  const _Float16* p = base + (size_t)m * ld + k0 + kh;
  v16h a;
#pragma unroll
  for (int i = 0; i < 8; ++i){ a[i] = p[i]; a[8+i] = p[8+i]; }
  return a;
}

// B fragment: 32(K) x 16(N) where B[k][n] = Wn[n*ld + k]  (N-major rows, K contiguous)
// lanes 0-15: K = k0+0..15 ; lanes 16-31: K = k0+16..31 ; lane%16 = N
__device__ __forceinline__ v16h ldB16_nrows(const _Float16* Wn, int ld, int k0, int lane){
  int n = lane & 15; int kh = (lane >> 4) << 4;
  const _Float16* p = Wn + (size_t)n * ld + k0 + kh;
  v16h b;
#pragma unroll
  for (int i = 0; i < 16; ++i) b[i] = p[i];
  return b;
}

// ---------------- one-time f32 -> f16 conversion ----------------------------------------
__global__ __launch_bounds__(256) void f32_to_f16_kernel(
    const float* __restrict__ in, _Float16* __restrict__ out, int n)
{
  int i = blockIdx.x * 256 + threadIdx.x;
  if (i < n) out[i] = (_Float16)in[i];
}

// ---------------- QKV projection: f16 [8192,1024] x f16 [3072,1024]^T -------------------
// async double-buffered LDS pipeline, scatter to head-major f16 q/k/v
__global__ __launch_bounds__(256) void qkv_gemm_kernel(
    const _Float16* __restrict__ X, const _Float16* __restrict__ W,
    _Float16* __restrict__ qh, _Float16* __restrict__ kh, _Float16* __restrict__ vh)
{
  __shared__ _Float16 Xs[2][128][40];
  __shared__ _Float16 Ws[2][128][40];
  const int tid = threadIdx.x;
  const int lane = tid & 31, wave = tid >> 5;
  const int bm = blockIdx.x * 128, bn = blockIdx.y * 128;
  const int wrow = (wave >> 2) * 64, wcol = (wave & 3) * 32;
  v8f acc[4][2];
#pragma unroll
  for (int i = 0; i < 4; ++i){ acc[i][0] = zero8(); acc[i][1] = zero8(); }
  const int r = tid >> 1, q = tid & 1;
  const _Float16* xrow = X + (size_t)(bm + r) * Cn + q * 16;
  const _Float16* wrow_p = W + (size_t)(bn + r) * Cn + q * 16;
  async_copy32((unsigned)(size_t)&Xs[0][r][q*16], xrow);
  async_copy32((unsigned)(size_t)&Ws[0][r][q*16], wrow_p);
  for (int it = 0; it < Cn/32; ++it){
    const int buf = it & 1;
    async_wait0();
    __syncthreads();
    if (it + 1 < Cn/32){
      async_copy32((unsigned)(size_t)&Xs[buf^1][r][q*16], xrow + (it+1)*32);
      async_copy32((unsigned)(size_t)&Ws[buf^1][r][q*16], wrow_p + (it+1)*32);
    }
    v16h bf0 = ldB16_nrows(&Ws[buf][wcol][0],      40, 0, lane);
    v16h bf1 = ldB16_nrows(&Ws[buf][wcol + 16][0], 40, 0, lane);
#pragma unroll
    for (int i4 = 0; i4 < 4; ++i4){
      v16h af = ldA16(&Xs[buf][wrow + i4*16][0], 40, 0, lane);
      acc[i4][0] = wmma16(af, bf0, acc[i4][0]);
      acc[i4][1] = wmma16(af, bf1, acc[i4][1]);
    }
  }
  const int n = lane & 15, mb = (lane >> 4) * 8;
#pragma unroll
  for (int i4 = 0; i4 < 4; ++i4)
#pragma unroll
    for (int j = 0; j < 2; ++j){
      int gn = bn + wcol + j*16 + n;
      int which = gn >> 10, o = gn & 1023, h = o >> 6, hs = o & 63;
      _Float16* dst = (which == 0) ? qh : ((which == 1) ? kh : vh);
#pragma unroll
      for (int rr = 0; rr < 8; ++rr){
        int gm = bm + wrow + i4*16 + mb + rr;
        int b = gm >> 10, t = gm & 1023;
        dst[(size_t)((b*Hn + h)*Tn + t) * HSn + hs] = (_Float16)acc[i4][j][rr];
      }
    }
}

// ---------------- Conv-compression as GEMM: patches[8064,2048] x conv_w^T[2048,64] ------
__global__ __launch_bounds__(32) void compress_gemm_kernel(
    const _Float16* __restrict__ kh, const _Float16* __restrict__ vh,
    const _Float16* __restrict__ convw, float* __restrict__ ck, float* __restrict__ cv)
{
  const _Float16* src = blockIdx.y ? vh : kh;
  float* dst = blockIdx.y ? cv : ck;
  const int lane = threadIdx.x & 31;
  const int tile = blockIdx.x;                  // 0..503, 16 rows each
  const int m = lane & 15;
  const int row = tile * 16 + m;                // 0..8063 = bh*63 + nb
  const int bh = row / NBn, nb = row % NBn;
  const _Float16* ab = src + (size_t)(bh * Tn + nb * 16) * HSn;  // patch base token
  const int khA = (lane >> 4) * 16;
  const int khB = (lane >> 4) * 16;
  const int nB  = lane & 15;
  v8f acc[4];
#pragma unroll
  for (int i = 0; i < 4; ++i) acc[i] = zero8();
  for (int k0 = 0; k0 < 2048; k0 += 32){
    v16h a;                                      // K index = hs_in*32 + kk
#pragma unroll
    for (int i = 0; i < 8; ++i){
      int k1 = k0 + khA + i;  a[i]   = ab[(size_t)(k1 & 31) * HSn + (k1 >> 5)];
      int k2 = k1 + 8;        a[8+i] = ab[(size_t)(k2 & 31) * HSn + (k2 >> 5)];
    }
#pragma unroll
    for (int nt = 0; nt < 4; ++nt){
      const _Float16* wp = convw + (size_t)(nt*16 + nB) * 2048 + k0 + khB;
      v16h b;
#pragma unroll
      for (int i = 0; i < 16; ++i) b[i] = wp[i];
      acc[nt] = wmma16(a, b, acc[nt]);
    }
  }
  const int mb = (lane >> 4) * 8;
#pragma unroll
  for (int nt = 0; nt < 4; ++nt)
#pragma unroll
    for (int rr = 0; rr < 8; ++rr)
      dst[(size_t)(tile*16 + mb + rr) * HSn + nt*16 + nB] = acc[nt][rr];
}

// ---------------- LN(HS) + exact GELU, emit f16 kc/vc padded to 64 blocks ---------------
__global__ __launch_bounds__(64) void comp_lngelu_kernel(
    const float* __restrict__ ck, const float* __restrict__ cv,
    const float* __restrict__ lnw, _Float16* __restrict__ kc, _Float16* __restrict__ vc)
{
  const float* srcb = blockIdx.y ? cv : ck;
  _Float16* dstb = blockIdx.y ? vc : kc;
  int row = blockIdx.x;                         // bh*64 + nb (nb==63 -> zero pad)
  int bh = row >> 6, nb = row & 63;
  int t = threadIdx.x;
  __shared__ float red[64];
  float val = (nb < NBn) ? srcb[(size_t)(bh * NBn + nb) * HSn + t] : 0.f;
  red[t] = val; __syncthreads();
  for (int s = 32; s > 0; s >>= 1){ if (t < s) red[t] += red[t+s]; __syncthreads(); }
  float mean = red[0] * (1.f/64.f); __syncthreads();
  float d = val - mean;
  red[t] = d * d; __syncthreads();
  for (int s = 32; s > 0; s >>= 1){ if (t < s) red[t] += red[t+s]; __syncthreads(); }
  float var = red[0] * (1.f/64.f);
  float y = d * rsqrtf(var + 1e-5f) * lnw[t];
  float g = 0.5f * y * (1.f + erff(y * 0.70710678118f));
  dstb[(size_t)row * HSn + t] = (nb < NBn) ? (_Float16)g : (_Float16)0.f;
}

// ---------------- Local banded attention: wave per 16-query tile ------------------------
__global__ __launch_bounds__(128) void local_attn_kernel(
    const _Float16* __restrict__ qh, const _Float16* __restrict__ kh,
    const _Float16* __restrict__ vh, float* __restrict__ lout)
{
  __shared__ float    Ssh[4][16][160];
  __shared__ _Float16 Psh[4][16][160];
  const int lane = threadIdx.x & 31, wave = threadIdx.x >> 5;
  const int qt = blockIdx.x * 4 + wave;
  const int bh = blockIdx.y;
  const int i0 = qt * 16;
  const _Float16* qb = qh + (size_t)(bh*Tn + i0) * HSn;
  const _Float16* kb = kh + (size_t)bh * Tn * HSn;
  const _Float16* vb = vh + (size_t)bh * Tn * HSn;
  float    (*S)[160] = Ssh[wave];
  _Float16 (*P)[160] = Psh[wave];
  const float NEG = -1.0e30f;
  const float scale = 0.125f;                   // 1/sqrt(64)
  v16h aq0 = ldA16(qb, HSn, 0,  lane);
  v16h aq1 = ldA16(qb, HSn, 32, lane);
  const int n = lane & 15, mb = (lane >> 4) * 8;
  for (int kt = 0; kt < 9; ++kt){
    int jt = qt - 8 + kt;                       // key tile (uniform per wave)
    if (jt < 0){
#pragma unroll
      for (int rr = 0; rr < 8; ++rr) S[mb+rr][kt*16+n] = NEG;
    } else {
      v8f s = zero8();
      const _Float16* kr = kb + (size_t)jt * 16 * HSn;
      s = wmma16(aq0, ldB16_nrows(kr, HSn, 0,  lane), s);
      s = wmma16(aq1, ldB16_nrows(kr, HSn, 32, lane), s);
#pragma unroll
      for (int rr = 0; rr < 8; ++rr){
        int i = i0 + mb + rr, j = jt*16 + n;
        float sv = s[rr] * scale;
        if (j > i || j + 128 < i) sv = NEG;     // band: i-128 <= j <= i
        S[mb+rr][kt*16+n] = sv;
      }
    }
  }
#pragma unroll
  for (int rr = 0; rr < 8; ++rr) S[mb+rr][144+n] = NEG;   // pad cols 144..159
  __syncthreads();
  for (int rr = 0; rr < 16; ++rr){
    float mx = NEG;
    for (int c = lane; c < 160; c += 32) mx = fmaxf(mx, S[rr][c]);
#pragma unroll
    for (int off = 16; off; off >>= 1) mx = fmaxf(mx, __shfl_xor(mx, off, 32));
    float sum = 0.f;
    for (int c = lane; c < 160; c += 32){ float e = __expf(S[rr][c]-mx); S[rr][c]=e; sum+=e; }
#pragma unroll
    for (int off = 16; off; off >>= 1) sum += __shfl_xor(sum, off, 32);
    float inv = 1.f / sum;
    for (int c = lane; c < 160; c += 32) P[rr][c] = (_Float16)(S[rr][c]*inv);
  }
  __syncthreads();
  v8f o[4];
#pragma unroll
  for (int nt = 0; nt < 4; ++nt) o[nt] = zero8();
  const int jbase = (qt - 8) * 16;
  const int khB = (lane >> 4) * 16;
  for (int ks = 0; ks < 5; ++ks){
    v16h a = ldA16(&P[0][0], 160, ks*32, lane);
#pragma unroll
    for (int nt = 0; nt < 4; ++nt){
      v16h b;
#pragma unroll
      for (int i = 0; i < 16; ++i){
        int j = jbase + ks*32 + khB + i;
        j = j < 0 ? 0 : (j > Tn-1 ? Tn-1 : j);  // clamped rows have P==0
        b[i] = vb[(size_t)j * HSn + nt*16 + n];
      }
      o[nt] = wmma16(a, b, o[nt]);
    }
  }
  int b0 = bh >> 4, h = bh & 15;
#pragma unroll
  for (int nt = 0; nt < 4; ++nt)
#pragma unroll
    for (int rr = 0; rr < 8; ++rr)
      lout[(size_t)(b0*Tn + i0 + mb + rr) * Cn + h*HSn + nt*16 + n] = o[nt][rr];
}

// ---------------- Compressed-KV attention (NB=63 padded to 64) --------------------------
__global__ __launch_bounds__(128) void comp_attn_kernel(
    const _Float16* __restrict__ qh, const _Float16* __restrict__ kc,
    const _Float16* __restrict__ vc, float* __restrict__ cout)
{
  __shared__ float    Ssh[4][16][64];
  __shared__ _Float16 Psh[4][16][64];
  const int lane = threadIdx.x & 31, wave = threadIdx.x >> 5;
  const int qt = blockIdx.x * 4 + wave;
  const int bh = blockIdx.y;
  const int i0 = qt * 16;
  const _Float16* qb  = qh + (size_t)(bh*Tn + i0) * HSn;
  const _Float16* kcb = kc + (size_t)bh * 64 * HSn;
  const _Float16* vcb = vc + (size_t)bh * 64 * HSn;
  float    (*S)[64] = Ssh[wave];
  _Float16 (*P)[64] = Psh[wave];
  const float NEG = -1.0e30f;
  const float scale = 0.125f;
  v16h aq0 = ldA16(qb, HSn, 0,  lane);
  v16h aq1 = ldA16(qb, HSn, 32, lane);
  const int n = lane & 15, mb = (lane >> 4) * 8;
  for (int kt = 0; kt < 4; ++kt){
    v8f s = zero8();
    const _Float16* kr = kcb + (size_t)kt * 16 * HSn;
    s = wmma16(aq0, ldB16_nrows(kr, HSn, 0,  lane), s);
    s = wmma16(aq1, ldB16_nrows(kr, HSn, 32, lane), s);
#pragma unroll
    for (int rr = 0; rr < 8; ++rr){
      int i = i0 + mb + rr, nbk = kt*16 + n;
      float sv = s[rr] * scale;
      if (nbk > i || nbk >= NBn) sv = NEG;      // block-causal + padding mask
      S[mb+rr][kt*16+n] = sv;
    }
  }
  __syncthreads();
  for (int rr = 0; rr < 16; ++rr){
    float mx = NEG;
    for (int c = lane; c < 64; c += 32) mx = fmaxf(mx, S[rr][c]);
#pragma unroll
    for (int off = 16; off; off >>= 1) mx = fmaxf(mx, __shfl_xor(mx, off, 32));
    float sum = 0.f;
    for (int c = lane; c < 64; c += 32){ float e = __expf(S[rr][c]-mx); S[rr][c]=e; sum+=e; }
#pragma unroll
    for (int off = 16; off; off >>= 1) sum += __shfl_xor(sum, off, 32);
    float inv = 1.f / sum;
    for (int c = lane; c < 64; c += 32) P[rr][c] = (_Float16)(S[rr][c]*inv);
  }
  __syncthreads();
  v8f o[4];
#pragma unroll
  for (int nt = 0; nt < 4; ++nt) o[nt] = zero8();
  const int khB = (lane >> 4) * 16;
  for (int ks = 0; ks < 2; ++ks){
    v16h a = ldA16(&P[0][0], 64, ks*32, lane);
#pragma unroll
    for (int nt = 0; nt < 4; ++nt){
      v16h b;
#pragma unroll
      for (int i = 0; i < 16; ++i)
        b[i] = vcb[(size_t)(ks*32 + khB + i) * HSn + nt*16 + n];
      o[nt] = wmma16(a, b, o[nt]);
    }
  }
  int b0 = bh >> 4, h = bh & 15;
#pragma unroll
  for (int nt = 0; nt < 4; ++nt)
#pragma unroll
    for (int rr = 0; rr < 8; ++rr)
      cout[(size_t)(b0*Tn + i0 + mb + rr) * Cn + h*HSn + nt*16 + n] = o[nt][rr];
}

// ---------------- LayerNorm over C=1024 rows --------------------------------------------
__global__ __launch_bounds__(256) void ln_rows_kernel(
    const float* __restrict__ X, const float* __restrict__ w, float* __restrict__ Y)
{
  int row = blockIdx.x, tid = threadIdx.x;
  __shared__ float red[256];
  const float* xr = X + (size_t)row * Cn;
  float s = 0.f, s2 = 0.f;
  for (int c = tid; c < Cn; c += 256){ float v = xr[c]; s += v; s2 += v*v; }
  red[tid] = s; __syncthreads();
  for (int st = 128; st > 0; st >>= 1){ if (tid < st) red[tid] += red[tid+st]; __syncthreads(); }
  float mean = red[0] * (1.f/Cn); __syncthreads();
  red[tid] = s2; __syncthreads();
  for (int st = 128; st > 0; st >>= 1){ if (tid < st) red[tid] += red[tid+st]; __syncthreads(); }
  float var = red[0] * (1.f/Cn) - mean*mean;
  float inv = rsqrtf(var + 1e-5f);
  for (int c = tid; c < Cn; c += 256)
    Y[(size_t)row * Cn + c] = (xr[c] - mean) * inv * w[c];
}

// ---------------- Gating MLP on last-token features (B=8, VALU) -------------------------
__global__ __launch_bounds__(256) void gating_kernel(
    const float* __restrict__ ln, const float* __restrict__ co,
    const float* __restrict__ w1, const float* __restrict__ glnw,
    const float* __restrict__ w2, float* __restrict__ gates)
{
  int b = blockIdx.x, tid = threadIdx.x;
  __shared__ float hbuf[Cn];
  __shared__ float red[256];
  const float* f1 = ln + ((size_t)b*Tn + (Tn-1)) * Cn;
  const float* f2 = co + ((size_t)b*Tn + (Tn-1)) * Cn;
  for (int c = tid; c < Cn; c += 256){
    const float* wr = w1 + (size_t)c * (2*Cn);
    float acc = 0.f;
    for (int j = 0; j < Cn; ++j) acc += f1[j] * wr[j];
    for (int j = 0; j < Cn; ++j) acc += f2[j] * wr[Cn + j];
    hbuf[c] = acc;
  }
  __syncthreads();
  float s = 0.f, s2 = 0.f;
  for (int c = tid; c < Cn; c += 256){ float v = hbuf[c]; s += v; s2 += v*v; }
  red[tid] = s; __syncthreads();
  for (int st = 128; st > 0; st >>= 1){ if (tid < st) red[tid] += red[tid+st]; __syncthreads(); }
  float mean = red[0] * (1.f/Cn); __syncthreads();
  red[tid] = s2; __syncthreads();
  for (int st = 128; st > 0; st >>= 1){ if (tid < st) red[tid] += red[tid+st]; __syncthreads(); }
  float var = red[0] * (1.f/Cn) - mean*mean;
  float inv = rsqrtf(var + 1e-5f);
  __syncthreads();
  for (int c = tid; c < Cn; c += 256){
    float v = (hbuf[c] - mean) * inv * glnw[c];
    hbuf[c] = v > 0.f ? v : 0.f;
  }
  __syncthreads();
  float l0 = 0.f, l1 = 0.f;
  for (int c = tid; c < Cn; c += 256){ l0 += hbuf[c]*w2[c]; l1 += hbuf[c]*w2[Cn+c]; }
  red[tid] = l0; __syncthreads();
  for (int st = 128; st > 0; st >>= 1){ if (tid < st) red[tid] += red[tid+st]; __syncthreads(); }
  l0 = red[0]; __syncthreads();
  red[tid] = l1; __syncthreads();
  for (int st = 128; st > 0; st >>= 1){ if (tid < st) red[tid] += red[tid+st]; __syncthreads(); }
  l1 = red[0];
  if (tid == 0){
    float mxl = fmaxf(l0, l1);
    float e0 = __expf(l0 - mxl), e1 = __expf(l1 - mxl);
    gates[b*2]   = e0 / (e0 + e1);
    gates[b*2+1] = e1 / (e0 + e1);
  }
}

// ---------------- Gated combine -> f16 --------------------------------------------------
__global__ __launch_bounds__(256) void combine_kernel(
    const float* __restrict__ ln, const float* __restrict__ co,
    const float* __restrict__ gates, _Float16* __restrict__ out)
{
  size_t idx = (size_t)blockIdx.x * 256 + threadIdx.x;
  if (idx < (size_t)Bn * Tn * Cn){
    int b = (int)(idx >> 20);                   // T*C == 1<<20
    out[idx] = (_Float16)(gates[2*b] * ln[idx] + gates[2*b+1] * co[idx]);
  }
}

// ---------------- Output projection: f16 [8192,1024] x f16 [1024,1024]^T -> f32 ---------
__global__ __launch_bounds__(256) void proj_gemm_kernel(
    const _Float16* __restrict__ X, const _Float16* __restrict__ W, float* __restrict__ Y)
{
  __shared__ _Float16 Xs[2][128][40];
  __shared__ _Float16 Ws[2][128][40];
  const int tid = threadIdx.x;
  const int lane = tid & 31, wave = tid >> 5;
  const int bm = blockIdx.x * 128, bn = blockIdx.y * 128;
  const int wrow = (wave >> 2) * 64, wcol = (wave & 3) * 32;
  v8f acc[4][2];
#pragma unroll
  for (int i = 0; i < 4; ++i){ acc[i][0] = zero8(); acc[i][1] = zero8(); }
  const int r = tid >> 1, q = tid & 1;
  const _Float16* xrow = X + (size_t)(bm + r) * Cn + q * 16;
  const _Float16* wrow_p = W + (size_t)(bn + r) * Cn + q * 16;
  async_copy32((unsigned)(size_t)&Xs[0][r][q*16], xrow);
  async_copy32((unsigned)(size_t)&Ws[0][r][q*16], wrow_p);
  for (int it = 0; it < Cn/32; ++it){
    const int buf = it & 1;
    async_wait0();
    __syncthreads();
    if (it + 1 < Cn/32){
      async_copy32((unsigned)(size_t)&Xs[buf^1][r][q*16], xrow + (it+1)*32);
      async_copy32((unsigned)(size_t)&Ws[buf^1][r][q*16], wrow_p + (it+1)*32);
    }
    v16h bf0 = ldB16_nrows(&Ws[buf][wcol][0],      40, 0, lane);
    v16h bf1 = ldB16_nrows(&Ws[buf][wcol + 16][0], 40, 0, lane);
#pragma unroll
    for (int i4 = 0; i4 < 4; ++i4){
      v16h af = ldA16(&Xs[buf][wrow + i4*16][0], 40, 0, lane);
      acc[i4][0] = wmma16(af, bf0, acc[i4][0]);
      acc[i4][1] = wmma16(af, bf1, acc[i4][1]);
    }
  }
  const int n = lane & 15, mb = (lane >> 4) * 8;
#pragma unroll
  for (int i4 = 0; i4 < 4; ++i4)
#pragma unroll
    for (int j = 0; j < 2; ++j){
      int gn = bn + wcol + j*16 + n;
#pragma unroll
      for (int rr = 0; rr < 8; ++rr){
        int gm = bm + wrow + i4*16 + mb + rr;
        Y[(size_t)gm * Cn + gn] = acc[i4][j][rr];
      }
    }
}

extern "C" void kernel_launch(void* const* d_in, const int* in_sizes, int n_in,
                              void* d_out, int out_size, void* d_ws, size_t ws_size,
                              hipStream_t stream)
{
  (void)in_sizes; (void)n_in; (void)out_size; (void)ws_size;
  const float* x     = (const float*)d_in[0];
  const float* cattn = (const float*)d_in[1];
  const float* convw = (const float*)d_in[2];
  const float* lncw  = (const float*)d_in[3];
  const float* lnlw  = (const float*)d_in[4];
  const float* gw1   = (const float*)d_in[5];
  const float* glnw  = (const float*)d_in[6];
  const float* gw2   = (const float*)d_in[7];
  const float* cproj = (const float*)d_in[8];
  float* y = (float*)d_out;

  char* wsb = (char*)d_ws;
  size_t off = 0;
  auto alloc = [&](size_t bytes) -> void* {
    void* p = wsb + off;
    off = (off + bytes + 255) & ~(size_t)255;
    return p;
  };
  const size_t NQ = (size_t)Bn * Hn * Tn * HSn;     // 8388608
  const int NX  = Bn * Tn * Cn;                     // 8388608
  const int NW1 = 3 * Cn * Cn;                      // 3145728
  const int NW2 = Cn * Cn;                          // 1048576
  const int NWC = HSn * HSn * 32;                   // 131072
  _Float16* qh   = (_Float16*)alloc(NQ * 2);
  _Float16* kh   = (_Float16*)alloc(NQ * 2);
  _Float16* vh   = (_Float16*)alloc(NQ * 2);
  float*    ck   = (float*)   alloc((size_t)BHn * NBn * HSn * 4);
  float*    cv   = (float*)   alloc((size_t)BHn * NBn * HSn * 4);
  _Float16* kc   = (_Float16*)alloc((size_t)BHn * 64 * HSn * 2);
  _Float16* vc   = (_Float16*)alloc((size_t)BHn * 64 * HSn * 2);
  float*    lout = (float*)   alloc((size_t)Bn * Tn * Cn * 4);
  float*    cout = (float*)   alloc((size_t)Bn * Tn * Cn * 4);
  float*    lnrm = (float*)   alloc((size_t)Bn * Tn * Cn * 4);
  _Float16* comb = (_Float16*)alloc((size_t)Bn * Tn * Cn * 2);
  float*    gts  = (float*)   alloc((size_t)Bn * 2 * 4);
  _Float16* xh     = (_Float16*)alloc((size_t)NX  * 2);
  _Float16* wqkvh  = (_Float16*)alloc((size_t)NW1 * 2);
  _Float16* wprojh = (_Float16*)alloc((size_t)NW2 * 2);
  _Float16* convwh = (_Float16*)alloc((size_t)NWC * 2);

  f32_to_f16_kernel   <<<(NX +255)/256, 256, 0, stream>>>(x,     xh,     NX);
  f32_to_f16_kernel   <<<(NW1+255)/256, 256, 0, stream>>>(cattn, wqkvh,  NW1);
  f32_to_f16_kernel   <<<(NW2+255)/256, 256, 0, stream>>>(cproj, wprojh, NW2);
  f32_to_f16_kernel   <<<(NWC+255)/256, 256, 0, stream>>>(convw, convwh, NWC);

  qkv_gemm_kernel     <<<dim3(64, 24), 256, 0, stream>>>(xh, wqkvh, qh, kh, vh);
  compress_gemm_kernel<<<dim3(504, 2),  32, 0, stream>>>(kh, vh, convwh, ck, cv);
  comp_lngelu_kernel  <<<dim3(BHn*64,2),64, 0, stream>>>(ck, cv, lncw, kc, vc);
  local_attn_kernel   <<<dim3(16, BHn),128, 0, stream>>>(qh, kh, vh, lout);
  comp_attn_kernel    <<<dim3(16, BHn),128, 0, stream>>>(qh, kc, vc, cout);
  ln_rows_kernel      <<<Bn*Tn,        256, 0, stream>>>(lout, lnlw, lnrm);
  gating_kernel       <<<Bn,           256, 0, stream>>>(lnrm, cout, gw1, glnw, gw2, gts);
  combine_kernel      <<<(Bn*Tn*Cn+255)/256, 256, 0, stream>>>(lnrm, cout, gts, comb);
  proj_gemm_kernel    <<<dim3(64, 8),  256, 0, stream>>>(comb, wprojh, y);
}
